// RelationalBlock_24953759990446
// MI455X (gfx1250) — compile-verified, matrix operand order checked by reading, not run on hardware
//
#include <hip/hip_runtime.h>

typedef unsigned short ushort_t;
typedef __attribute__((ext_vector_type(16))) __bf16 v16bf;
typedef __attribute__((ext_vector_type(8)))  float  v8f;

union Frag {
    v16bf v;
    uint4 q[2];
    ushort_t s[16];
};

__device__ inline ushort_t f2bf(float f) {
    unsigned int u = __float_as_uint(f);
    unsigned int r = (u + 0x7FFFu + ((u >> 16) & 1u)) >> 16;
    return (ushort_t)r;
}
__device__ inline float bf2f(ushort_t h) {
    return __uint_as_float(((unsigned int)h) << 16);
}

// ---------------------------------------------------------------------------
// gfx1250 async global->LDS DMA (ASYNCcnt-tracked).  LDS destination offset
// is the low 32 bits of the flat shared-pointer (ISA 10.2: LDS aperture
// addresses carry the wave-relative LDS offset in addr[31:0]).
// ---------------------------------------------------------------------------
__device__ inline unsigned lds_off_of(const void* p) {
    return (unsigned)(size_t)p;
}
__device__ inline void async_load_b128(unsigned lds_off, const void* gaddr) {
    asm volatile("global_load_async_to_lds_b128 %0, %1, off"
                 :: "v"(lds_off), "v"((unsigned long long)(size_t)gaddr)
                 : "memory");
}
#define WAIT_ASYNCCNT(n) asm volatile("s_wait_asynccnt " #n ::: "memory")

// ---------------------------------------------------------------------------
// Weight cast fp32 -> bf16 (grid-stride)
// ---------------------------------------------------------------------------
__global__ void cast_f32_bf16_kernel(const float* __restrict__ src,
                                     ushort_t* __restrict__ dst, int n) {
    for (int i = blockIdx.x * blockDim.x + threadIdx.x; i < n;
         i += gridDim.x * blockDim.x)
        dst[i] = f2bf(src[i]);
}

// ---------------------------------------------------------------------------
// RMSNorm (fp32 in) -> bf16 out.  One row (D=1024) per 256-thread block.
// ---------------------------------------------------------------------------
__global__ __launch_bounds__(256) void rmsnorm_cast_kernel(
    const float* __restrict__ x, const float* __restrict__ w,
    ushort_t* __restrict__ out) {
    const int D = 1024;
    int row = blockIdx.x;
    int t = threadIdx.x;
    const float4* xr = (const float4*)(x + (size_t)row * D);
    float4 v = xr[t];
    float ss = v.x * v.x + v.y * v.y + v.z * v.z + v.w * v.w;
#pragma unroll
    for (int m = 16; m >= 1; m >>= 1) ss += __shfl_xor(ss, m, 32);
    __shared__ float red[8];
    if ((t & 31) == 0) red[t >> 5] = ss;
    __syncthreads();
    float tot = 0.f;
#pragma unroll
    for (int i = 0; i < 8; ++i) tot += red[i];
    float scale = rsqrtf(tot * (1.0f / 1024.0f) + 1.1920929e-07f);
    const float4* wr = (const float4*)w;
    float4 wv = wr[t];
    unsigned int a0 = (unsigned int)f2bf(v.x * scale * wv.x) |
                      ((unsigned int)f2bf(v.y * scale * wv.y) << 16);
    unsigned int a1 = (unsigned int)f2bf(v.z * scale * wv.z) |
                      ((unsigned int)f2bf(v.w * scale * wv.w) << 16);
    uint2 o; o.x = a0; o.y = a1;
    ((uint2*)(out + (size_t)row * D))[t] = o;
}

// ---------------------------------------------------------------------------
// SiLU(g) * u, bf16 in/out
// ---------------------------------------------------------------------------
__global__ void silu_mul_kernel(const ushort_t* __restrict__ g,
                                const ushort_t* __restrict__ u,
                                ushort_t* __restrict__ t, int n) {
    int i = blockIdx.x * blockDim.x + threadIdx.x;
    if (i < n) {
        float gv = bf2f(g[i]);
        float uv = bf2f(u[i]);
        float s = gv / (1.f + __expf(-gv));
        t[i] = f2bf(s * uv);
    }
}

// ---------------------------------------------------------------------------
// bf16 WMMA GEMM:  C[.,z] = A * B[z]   (bf16 row-major operands)
// 128x128 macro-tile / block, 256 threads = 8 waves, each wave 64x32.
// blockIdx.z selects one of up to 3 weight/output pairs (QKV / gate+up).
// A tile: DOUBLE-BUFFERED async global->LDS DMA (2 x b128 per thread per
//   K-step); tile k+1's DMA runs under tile k's WMMAs, s_wait_asynccnt 2
//   guarantees residency (async loads complete in order).
// B tile: register-pipelined global loads + fragment-order LDS scatter.
// All fragment reads are contiguous 32B ds loads.
// ---------------------------------------------------------------------------
template <int WRITE_F32, int RESID>
__global__ __launch_bounds__(256) void gemm_bf16_kernel(
    const ushort_t* __restrict__ A,
    const ushort_t* __restrict__ B0, const ushort_t* __restrict__ B1,
    const ushort_t* __restrict__ B2,
    float* Cf,
    ushort_t* Cb0, ushort_t* Cb1, ushort_t* Cb2,
    const float* res, int M, int N, int K) {
    __shared__ alignas(32) ushort_t Af[2][8][32][16];   // 16 KB, ping-pong
    __shared__ alignas(32) ushort_t Bf[8][32][16];      // 8 KB

    const ushort_t* B = (blockIdx.z == 0) ? B0 : (blockIdx.z == 1 ? B1 : B2);
    ushort_t* Cb = (blockIdx.z == 0) ? Cb0 : (blockIdx.z == 1 ? Cb1 : Cb2);

    int tid = threadIdx.x;
    int mBase = blockIdx.y * 128;
    int nBase = blockIdx.x * 128;
    int wave = tid >> 5, lid = tid & 31;
    int wm = wave >> 2, wn = wave & 3;

    v8f acc[4][2] = {};

    // A-tile DMA mapping (each thread: 16 bf16 of one row -> two frag slots)
    int am = tid >> 1;
    int akseg = (tid & 1) << 4;
    int asub = am >> 4, alane = am & 15, aeoff = (akseg >> 4) << 3;
    unsigned ldsA[2][2];
#pragma unroll
    for (int bfr = 0; bfr < 2; ++bfr) {
        ldsA[bfr][0] = lds_off_of(&Af[bfr][asub][alane][aeoff]);
        ldsA[bfr][1] = lds_off_of(&Af[bfr][asub][alane + 16][aeoff]);
    }
    // B-tile loader mapping (each thread: 16 bf16 of one k-row)
    int bk = tid >> 3;
    int bnseg = (tid & 7) << 4;
    int bsub = tid & 7;
    int bhalf = (bk & 15) >> 3;
    int be = ((bk >> 4) << 3) + (bk & 7);

    const ushort_t* Aptr = A + (size_t)(mBase + am) * K + akseg;
    const ushort_t* Bptr = B + (size_t)bk * N + nBase + bnseg;
    int nk = K >> 5;

    // prologue: DMA tile 0 into buffer 0, B tile 0 into registers
    async_load_b128(ldsA[0][0], Aptr);
    async_load_b128(ldsA[0][1], Aptr + 8);
    uint4 b0 = *(const uint4*)(Bptr);
    uint4 b1 = *(const uint4*)(Bptr + 8);

    for (int kb = 0; kb < nk; ++kb) {
        int buf = kb & 1;
        // stage B tile: 16-way b16 scatter into fragment order
        {
            const ushort_t* pb0 = (const ushort_t*)&b0;
            const ushort_t* pb1 = (const ushort_t*)&b1;
#pragma unroll
            for (int nn = 0; nn < 8; ++nn) {
                Bf[bsub][nn + 16 * bhalf][be] = pb0[nn];
                Bf[bsub][nn + 8 + 16 * bhalf][be] = pb1[nn];
            }
        }
        bool more = (kb + 1 < nk);
        if (more) {
            // kick off next A tile DMA into the other buffer (that buffer's
            // readers finished at the previous iteration's trailing barrier)
            Aptr += 32;
            async_load_b128(ldsA[buf ^ 1][0], Aptr);
            async_load_b128(ldsA[buf ^ 1][1], Aptr + 8);
            // next B tile into registers (overlaps WMMAs below)
            Bptr += (size_t)32 * N;
            b0 = *(const uint4*)(Bptr);
            b1 = *(const uint4*)(Bptr + 8);
            __builtin_prefetch(Bptr + (size_t)32 * N, 0, 3);
            WAIT_ASYNCCNT(0x2);          // current tile resident, next in flight
        } else {
            WAIT_ASYNCCNT(0x0);
        }
        __syncthreads();

        Frag av[4], bv[2];
#pragma unroll
        for (int i = 0; i < 4; ++i)
            av[i].v = *(const v16bf*)&Af[buf][wm * 4 + i][lid][0];
#pragma unroll
        for (int j = 0; j < 2; ++j)
            bv[j].v = *(const v16bf*)&Bf[wn * 2 + j][lid][0];
#pragma unroll
        for (int i = 0; i < 4; ++i)
#pragma unroll
            for (int j = 0; j < 2; ++j)
                acc[i][j] = __builtin_amdgcn_wmma_f32_16x16x32_bf16(
                    false, av[i].v, false, bv[j].v, (short)0, acc[i][j],
                    false, false);
        __syncthreads();
    }

    int rbase = mBase + wm * 64 + ((lid >> 4) << 3);
    int cbase = nBase + wn * 32 + (lid & 15);
#pragma unroll
    for (int i = 0; i < 4; ++i)
#pragma unroll
        for (int j = 0; j < 2; ++j)
#pragma unroll
            for (int r = 0; r < 8; ++r) {
                int row = rbase + i * 16 + r;
                int col = cbase + j * 16;
                size_t idx = (size_t)row * N + col;
                float v = acc[i][j][r];
                if (RESID) v += res[idx];
                if (WRITE_F32) Cf[idx] = v;
                else           Cb[idx] = f2bf(v);
            }
}

// ---------------------------------------------------------------------------
// Flash attention, one wave per 16-query tile (4 per block), online softmax,
// analytic mask.  Q/K/V/O are bf16 [B,S,D] with head columns h*64..h*64+63.
// The 32-key V tile is DMA'd into LDS once per BLOCK via async-to-LDS; the
// DMA runs under the softmax and is drained with s_wait_asynccnt 0 right
// before the consuming barrier.  K B-fragments are hd-contiguous -> direct
// b128 global loads.
// MT: 0=col (q%32==k%32), 1=feat (q/128==k/128), 2=nbr (|q-k|<=64), 3=full
// ---------------------------------------------------------------------------
template <int MT>
__global__ __launch_bounds__(128) void flash_attn_kernel(
    const ushort_t* __restrict__ Q, const ushort_t* __restrict__ K,
    const ushort_t* __restrict__ V, ushort_t* __restrict__ O) {
    const int S = 1024, D = 1024, HD = 64;
    int tid = threadIdx.x;
    int lane = tid & 31;
    int wave = tid >> 5;
    int qtile = blockIdx.x * 4 + wave;
    int h = blockIdx.y;
    int b = blockIdx.z;
    int qBase = qtile * 16;
    size_t bS = (size_t)b * S;
    const ushort_t* Qb = Q + bS * D + (size_t)h * HD;
    const ushort_t* Kb = K + bS * D + (size_t)h * HD;
    const ushort_t* Vb = V + bS * D + (size_t)h * HD;

    __shared__ alignas(32) ushort_t Pf[4][32][16];   // probs, A-frag layout
    __shared__ alignas(16) ushort_t Vl[32][64];      // shared 32-key V tile

    int hbase = (lane >> 4) << 3;
    int myRowOff = hbase;                  // rows r + 8*(lane/16)

    // V-stager mapping: 128 threads x 16 bf16 (two async b128 each)
    int vkey = tid >> 2;
    int vseg = (tid & 3) << 4;
    unsigned ldsV0 = lds_off_of(&Vl[vkey][vseg]);
    unsigned ldsV1 = lds_off_of(&Vl[vkey][vseg + 8]);

    // Q A-fragments (hd chunks of 32), straight from global
    Frag qa[2];
    {
        int qrow = qBase + (lane & 15);
#pragma unroll
        for (int c = 0; c < 2; ++c) {
            const ushort_t* p = Qb + (size_t)qrow * D + c * 32 + hbase;
            qa[c].q[0] = *(const uint4*)(p);
            qa[c].q[1] = *(const uint4*)(p + 16);
        }
    }

    v8f o[4] = {};
    float mr[8], sr[8];
#pragma unroll
    for (int r = 0; r < 8; ++r) { mr[r] = -1e30f; sr[r] = 0.f; }
    const float scale = 0.125f;            // 1/sqrt(64)

    for (int kp = 0; kp < 32; ++kp) {      // 32 keys per step
        int kBase = kp * 32;

        // scores: Q(16x64) x K^T(64x32) via 4 WMMAs
        v8f sc[2] = {};
#pragma unroll
        for (int j = 0; j < 2; ++j)
#pragma unroll
            for (int c = 0; c < 2; ++c) {
                Frag kf;
                int key = kBase + j * 16 + (lane & 15);
                const ushort_t* p = Kb + (size_t)key * D + c * 32 + hbase;
                kf.q[0] = *(const uint4*)(p);
                kf.q[1] = *(const uint4*)(p + 16);
                sc[j] = __builtin_amdgcn_wmma_f32_16x16x32_bf16(
                    false, qa[c].v, false, kf.v, (short)0, sc[j], false, false);
            }

        __syncthreads();   // previous step's V consumers are done
        // async DMA of the V tile (32 x 64) -> LDS, overlaps the softmax
        {
            const ushort_t* p = Vb + (size_t)(kBase + vkey) * D + vseg;
            async_load_b128(ldsV0, p);
            async_load_b128(ldsV1, p + 8);
        }

        // mask + scale + online softmax
        int kc = kBase + (lane & 15);
        float p0[8], p1[8];
#pragma unroll
        for (int r = 0; r < 8; ++r) {
            int qr = qBase + myRowOff + r;
            float s0 = sc[0][r] * scale;
            float s1 = sc[1][r] * scale;
            int k0 = kc, k1 = kc + 16;
            bool m0, m1;
            if (MT == 0) { m0 = ((qr & 31) == (k0 & 31)); m1 = ((qr & 31) == (k1 & 31)); }
            else if (MT == 1) { m0 = ((qr >> 7) == (k0 >> 7)); m1 = ((qr >> 7) == (k1 >> 7)); }
            else if (MT == 2) { m0 = (abs(qr - k0) <= 64); m1 = (abs(qr - k1) <= 64); }
            else { m0 = true; m1 = true; }
            if (!m0) s0 = -1e30f;
            if (!m1) s1 = -1e30f;
            float mx = fmaxf(s0, s1);
#pragma unroll
            for (int d = 1; d < 16; d <<= 1) mx = fmaxf(mx, __shfl_xor(mx, d, 16));
            float mnew = fmaxf(mr[r], mx);
            float alpha = __expf(mr[r] - mnew);
            p0[r] = __expf(s0 - mnew);
            p1[r] = __expf(s1 - mnew);
            float rs = p0[r] + p1[r];
#pragma unroll
            for (int d = 1; d < 16; d <<= 1) rs += __shfl_xor(rs, d, 16);
            sr[r] = sr[r] * alpha + rs;
            mr[r] = mnew;
#pragma unroll
            for (int t = 0; t < 4; ++t) o[t][r] *= alpha;
        }

        // scatter probs (C layout) -> A-fragment layout in LDS, bf16
        {
            int phalf = (lane & 15) >> 3;
            int pe = lane & 7;
#pragma unroll
            for (int r = 0; r < 8; ++r) {
                int rl = myRowOff + r;
                Pf[wave][rl + 16 * phalf][pe] = f2bf(p0[r]);
                Pf[wave][rl + 16 * phalf][8 + pe] = f2bf(p1[r]);
            }
        }
        Frag pa;
        pa.v = *(const v16bf*)&Pf[wave][lane][0];

        WAIT_ASYNCCNT(0x0);   // V DMA landed
        __syncthreads();      // visible to all waves

        // O += P(16x32) x V(32x64): 4 WMMAs over hd-column tiles
#pragma unroll
        for (int t = 0; t < 4; ++t) {
            Frag vf;
            int n = t * 16 + (lane & 15);
#pragma unroll
            for (int e = 0; e < 16; ++e) {
                int kk = ((e >> 3) << 4) + ((lane >> 4) << 3) + (e & 7);
                vf.s[e] = Vl[kk][n];
            }
            o[t] = __builtin_amdgcn_wmma_f32_16x16x32_bf16(
                false, pa.v, false, vf.v, (short)0, o[t], false, false);
        }
    }

    // normalize + write bf16
#pragma unroll
    for (int t = 0; t < 4; ++t)
#pragma unroll
        for (int r = 0; r < 8; ++r) {
            int qr = qBase + myRowOff + r;
            int col = t * 16 + (lane & 15);
            O[(bS + qr) * D + h * HD + col] = f2bf(o[t][r] / sr[r]);
        }
}

// ---------------------------------------------------------------------------
// Host orchestration
// ---------------------------------------------------------------------------
extern "C" void kernel_launch(void* const* d_in, const int* in_sizes, int n_in,
                              void* d_out, int out_size, void* d_ws,
                              size_t ws_size, hipStream_t stream) {
    (void)in_sizes; (void)n_in; (void)out_size; (void)ws_size;
    const int B = 2, S = 1024, D = 1024, DFF = 4096;
    const int M = B * S;                   // 2048 rows
    const size_t DD = (size_t)D * D;
    const size_t BSD = (size_t)M * D;
    const size_t BSF = (size_t)M * DFF;

    const float* x = (const float*)d_in[0];
    const float* norms[5] = {(const float*)d_in[5], (const float*)d_in[10],
                             (const float*)d_in[15], (const float*)d_in[20],
                             (const float*)d_in[25]};
    // weight input indices: wq/wk/wv/wo x4 layers, then w_up, w_gate, w_down
    const int widx[19] = {6, 7, 8, 9, 11, 12, 13, 14, 16, 17, 18, 19,
                          21, 22, 23, 24, 26, 27, 28};
    size_t wsz[19];
    for (int i = 0; i < 16; ++i) wsz[i] = DD;
    wsz[16] = (size_t)D * DFF;   // w_up
    wsz[17] = (size_t)D * DFF;   // w_gate
    wsz[18] = (size_t)DFF * D;   // w_down

    ushort_t* ws = (ushort_t*)d_ws;
    size_t off = 0;
    ushort_t* wbf[19];
    for (int i = 0; i < 19; ++i) { wbf[i] = ws + off; off += wsz[i]; }
    ushort_t* XN  = ws + off; off += BSD;
    ushort_t* Qb  = ws + off; off += BSD;
    ushort_t* Kb  = ws + off; off += BSD;
    ushort_t* Vb  = ws + off; off += BSD;
    ushort_t* ATT = ws + off; off += BSD;
    ushort_t* G   = ws + off; off += BSF;
    ushort_t* U   = ws + off; off += BSF;
    ushort_t* T   = ws + off; off += BSF;

    // 1) cast weights to bf16
    for (int i = 0; i < 19; ++i)
        cast_f32_bf16_kernel<<<512, 256, 0, stream>>>(
            (const float*)d_in[widx[i]], wbf[i], (int)wsz[i]);

    // 2) residual stream lives in d_out (fp32)
    float* xf = (float*)d_out;
    hipMemcpyAsync(xf, x, BSD * sizeof(float), hipMemcpyDeviceToDevice, stream);

    dim3 gD(D / 128, M / 128, 1);      // N=1024, single output
    dim3 gQKV(D / 128, M / 128, 3);    // fused Q,K,V
    dim3 gGU(DFF / 128, M / 128, 2);   // fused gate,up
    dim3 gFA(S / 64, 16, B);           // flash: 4 qtiles per block

    for (int L = 0; L < 4; ++L) {
        rmsnorm_cast_kernel<<<M, 256, 0, stream>>>(xf, norms[L], XN);
        gemm_bf16_kernel<0, 0><<<gQKV, 256, 0, stream>>>(
            XN, wbf[L * 4 + 0], wbf[L * 4 + 1], wbf[L * 4 + 2],
            nullptr, Qb, Kb, Vb, nullptr, M, D, D);
        switch (L) {
        case 0: flash_attn_kernel<0><<<gFA, 128, 0, stream>>>(Qb, Kb, Vb, ATT); break;
        case 1: flash_attn_kernel<1><<<gFA, 128, 0, stream>>>(Qb, Kb, Vb, ATT); break;
        case 2: flash_attn_kernel<2><<<gFA, 128, 0, stream>>>(Qb, Kb, Vb, ATT); break;
        default: flash_attn_kernel<3><<<gFA, 128, 0, stream>>>(Qb, Kb, Vb, ATT); break;
        }
        gemm_bf16_kernel<1, 1><<<gD, 256, 0, stream>>>(
            ATT, wbf[L * 4 + 3], wbf[L * 4 + 3], wbf[L * 4 + 3],
            xf, nullptr, nullptr, nullptr, xf, M, D, D);
    }

    // FFN: x += (silu(h @ w_gate) * (h @ w_up)) @ w_down
    rmsnorm_cast_kernel<<<M, 256, 0, stream>>>(xf, norms[4], XN);
    gemm_bf16_kernel<0, 0><<<gGU, 256, 0, stream>>>(
        XN, wbf[17], wbf[16], wbf[16],
        nullptr, G, U, U, nullptr, M, DFF, D);          // gate (z=0), up (z=1)
    {
        int n = (int)BSF;
        silu_mul_kernel<<<(n + 255) / 256, 256, 0, stream>>>(G, U, T, n);
    }
    gemm_bf16_kernel<1, 1><<<gD, 256, 0, stream>>>(
        T, wbf[18], wbf[18], wbf[18],
        xf, nullptr, nullptr, nullptr, xf, M, D, DFF);  // down + residual
}
